// SAGATLayer_575525618146
// MI455X (gfx1250) — compile-verified, matrix operand order checked by reading, not run on hardware
//
#include <hip/hip_runtime.h>
#include <hip/hip_bf16.h>
#include <math.h>

// ---------------------------------------------------------------------------
// SAGATLayer for MI455X (gfx1250): WMMA bf16 node transforms (pre-converted
// bf16 operands, transposed W for contiguous B fragments) + LDS-staged
// atomic segment-softmax GATv2 edge passes.
// ---------------------------------------------------------------------------

#define IN_DIM   128
#define C_DIM    32
#define H_ALL    4          // 3 std heads + 1 rep head
#define EE_HID   32
#define EE_OUT   16
#define NEG_SLOPE 0.2f

typedef __bf16  bf16x16 __attribute__((ext_vector_type(16)));
typedef float   f32x8   __attribute__((ext_vector_type(8)));

// ---------------------------------------------------------------------------
// float atomic max via int/uint trick (lowers to global_atomic_max/min)
// ---------------------------------------------------------------------------
__device__ __forceinline__ void atomicMaxFloat(float* addr, float val) {
    if (val >= 0.0f) {
        atomicMax((int*)addr, __float_as_int(val));
    } else {
        atomicMin((unsigned int*)addr, (unsigned int)__float_as_int(val));
    }
}

// ---------------------------------------------------------------------------
// Pre-convert kernels: f32 -> bf16 (x row-major; W transposed to [Mout, K])
// ---------------------------------------------------------------------------
__global__ void cvt_x_bf16(const float* __restrict__ X, __bf16* __restrict__ Xb,
                           long long n) {
    const long long stride = (long long)gridDim.x * blockDim.x;
    for (long long i = (long long)blockIdx.x * blockDim.x + threadIdx.x; i < n; i += stride)
        Xb[i] = (__bf16)X[i];
}

__global__ void cvt_w_bf16_t(const float* __restrict__ W,   // [K, Mout]
                             __bf16* __restrict__ Wt,       // [Mout, K]
                             int K, int Mout) {
    const int total = K * Mout;
    const int stride = gridDim.x * blockDim.x;
    for (int i = blockIdx.x * blockDim.x + threadIdx.x; i < total; i += stride) {
        const int k = i / Mout, m = i - k * Mout;
        Wt[m * K + k] = (__bf16)W[i];
    }
}

// ---------------------------------------------------------------------------
// K1: Y[:, colOff:colOff+Mout] = X[N,128] @ W[128,Mout] + bias  (bf16 WMMA)
// One wave computes one 16x16 tile with 4 chained v_wmma_f32_16x16x32_bf16.
// A fragment: contiguous bf16 pairs from Xb row; B fragment: contiguous bf16
// pairs from transposed Wt row (column of W).
// ---------------------------------------------------------------------------
__global__ __launch_bounds__(32)
void gemm_wmma_bf16(const __bf16* __restrict__ Xb,   // [N, 128]
                    const __bf16* __restrict__ Wt,   // [Mout, 128]
                    const float* __restrict__ bias,
                    float* __restrict__ Y,
                    int N, int Mout, int ldY, int colOff) {
    const int tm   = blockIdx.x * 16;          // row tile base
    const int tn   = blockIdx.y * 16;          // col tile base (within Mout)
    const int lane = threadIdx.x;              // 0..31
    const int r16  = lane & 15;
    const int half = lane >> 4;                // 0: lanes 0-15, 1: lanes 16-31

    const int arow = min(tm + r16, N - 1);     // clamp for safety
    const __bf16* xrow = Xb + (size_t)arow * IN_DIM;
    const __bf16* wrow = Wt + (size_t)(tn + r16) * IN_DIM;

    f32x8 acc = {};
    #pragma unroll
    for (int k0 = 0; k0 < IN_DIM; k0 += 32) {
        bf16x16 a, b;
        // 16-bit A (16x32) / B (32x16) lane layout: VGPR v<4 -> K = 2v + 8*half,
        // VGPR v>=4 -> K = 16 + 2(v-4) + 8*half; pairs (K, K+1) contiguous.
        #pragma unroll
        for (int v = 0; v < 8; ++v) {
            const int kk = (v < 4) ? (2 * v + 8 * half)
                                   : (16 + 2 * (v - 4) + 8 * half);
            a[2 * v]     = xrow[k0 + kk];
            a[2 * v + 1] = xrow[k0 + kk + 1];
            b[2 * v]     = wrow[k0 + kk];
            b[2 * v + 1] = wrow[k0 + kk + 1];
        }
        acc = __builtin_amdgcn_wmma_f32_16x16x32_bf16(
                  false, a, false, b, (short)0, acc, false, false);
    }

    const float bv = bias[tn + r16];
    // D layout: VGPR r -> row = r + 8*half, col = lane&15
    #pragma unroll
    for (int r = 0; r < 8; ++r) {
        const int orow = tm + r + 8 * half;
        if (orow < N)
            Y[(size_t)orow * ldY + colOff + tn + r16] = acc[r] + bv;
    }
}

// ---------------------------------------------------------------------------
// K0: init out <- bias, smax <- -inf, denom <- 0, mean accumulators <- 0
// ---------------------------------------------------------------------------
__global__ void init_kernel(float* __restrict__ out,
                            const float* __restrict__ bias_s,
                            const float* __restrict__ bias_r,
                            float* __restrict__ smax,
                            float* __restrict__ denom,
                            float* __restrict__ accum,   // 16+16+1 floats
                            int N) {
    const int tid = blockIdx.x * blockDim.x + threadIdx.x;
    if (tid < 33) accum[tid] = 0.0f;
    for (int n = tid; n < N; n += gridDim.x * blockDim.x) {
        float* orow = out + (size_t)n * (H_ALL * C_DIM);
        #pragma unroll
        for (int j = 0; j < 96; ++j)  orow[j]      = bias_s[j];
        #pragma unroll
        for (int j = 0; j < 32; ++j)  orow[96 + j] = bias_r[j];
        #pragma unroll
        for (int h = 0; h < H_ALL; ++h) {
            smax[(size_t)n * H_ALL + h]  = -INFINITY;
            denom[(size_t)n * H_ALL + h] = 0.0f;
        }
    }
}

// ---------------------------------------------------------------------------
// K2: edge encoder MLP 7->32->16 (weights in LDS), store ea, mean sums
// ---------------------------------------------------------------------------
__global__ void edge_encoder_kernel(const float* __restrict__ raw,
                                    const float* __restrict__ w1,  // [7,32]
                                    const float* __restrict__ b1,  // [32]
                                    const float* __restrict__ w2,  // [32,16]
                                    const float* __restrict__ b2,  // [16]
                                    const unsigned char* __restrict__ mask,
                                    float* __restrict__ ea,
                                    float* __restrict__ accum, // [0:16) sum, [16:32) sum_rep, [32] cnt
                                    long long E) {
    __shared__ float s_w1[7 * EE_HID];
    __shared__ float s_b1[EE_HID];
    __shared__ float s_w2[EE_HID * EE_OUT];
    __shared__ float s_b2[EE_OUT];
    __shared__ float s_sum[EE_OUT];
    __shared__ float s_sum_rep[EE_OUT];
    __shared__ float s_cnt;
    for (int i = threadIdx.x; i < 7 * EE_HID; i += blockDim.x)     s_w1[i] = w1[i];
    for (int i = threadIdx.x; i < EE_HID; i += blockDim.x)         s_b1[i] = b1[i];
    for (int i = threadIdx.x; i < EE_HID * EE_OUT; i += blockDim.x) s_w2[i] = w2[i];
    for (int i = threadIdx.x; i < EE_OUT; i += blockDim.x) {
        s_b2[i] = b2[i]; s_sum[i] = 0.0f; s_sum_rep[i] = 0.0f;
    }
    if (threadIdx.x == 0) s_cnt = 0.0f;
    __syncthreads();

    const long long stride = (long long)gridDim.x * blockDim.x;
    for (long long e = (long long)blockIdx.x * blockDim.x + threadIdx.x; e < E; e += stride) {
        float r[7];
        #pragma unroll
        for (int i = 0; i < 7; ++i) r[i] = raw[e * 7 + i];

        float hid[EE_HID];
        #pragma unroll
        for (int j = 0; j < EE_HID; ++j) {
            float v = s_b1[j];
            #pragma unroll
            for (int i = 0; i < 7; ++i) v += r[i] * s_w1[i * EE_HID + j];
            hid[j] = v > 0.0f ? v : 0.0f;
        }
        const bool m = mask[e] != 0;
        #pragma unroll
        for (int j = 0; j < EE_OUT; ++j) {
            float v = s_b2[j];
            #pragma unroll
            for (int i = 0; i < EE_HID; ++i) v += hid[i] * s_w2[i * EE_OUT + j];
            ea[e * EE_OUT + j] = v;
            atomicAdd(&s_sum[j], v);
            if (m) atomicAdd(&s_sum_rep[j], v);
        }
        if (m) atomicAdd(&s_cnt, 1.0f);
    }
    __syncthreads();
    if (threadIdx.x < EE_OUT) {
        atomicAdd(&accum[threadIdx.x], s_sum[threadIdx.x]);
        atomicAdd(&accum[EE_OUT + threadIdx.x], s_sum_rep[threadIdx.x]);
    }
    if (threadIdx.x == 0) atomicAdd(&accum[32], s_cnt);
}

// ---------------------------------------------------------------------------
// K3: finalize means, precompute self-loop ef vectors (96 std + 32 rep)
// ---------------------------------------------------------------------------
__global__ void finalize_means_kernel(const float* __restrict__ accum,
                                      const float* __restrict__ we_s, // [16,96]
                                      const float* __restrict__ we_r, // [16,32]
                                      float* __restrict__ ef_loop,    // [128]
                                      long long E) {
    __shared__ float mean[EE_OUT];
    __shared__ float mrep[EE_OUT];
    if (threadIdx.x < EE_OUT) {
        mean[threadIdx.x] = accum[threadIdx.x] / (float)E;
        const float cnt = accum[32];
        mrep[threadIdx.x] = accum[EE_OUT + threadIdx.x] / fmaxf(cnt, 1.0f);
    }
    __syncthreads();
    const int t = threadIdx.x;
    if (t < 96) {
        float v = 0.0f;
        #pragma unroll
        for (int j = 0; j < EE_OUT; ++j) v += mean[j] * we_s[j * 96 + t];
        ef_loop[t] = v;
    }
    if (t < 32) {
        float v = 0.0f;
        #pragma unroll
        for (int j = 0; j < EE_OUT; ++j) v += mrep[j] * we_r[j * 32 + t];
        ef_loop[96 + t] = v;
    }
}

// ---------------------------------------------------------------------------
// K4: per-(edge,head) raw score + atomic segment max over dst.
// Small uniform weights staged in LDS; xl/xr gathered as float4 (b128).
// ---------------------------------------------------------------------------
__global__ void edge_score_kernel(const int* __restrict__ src,
                                  const int* __restrict__ dst,
                                  const unsigned char* __restrict__ mask,
                                  const float* __restrict__ ea,       // [E,16]
                                  const float* __restrict__ we_s,     // [16,96]
                                  const float* __restrict__ we_r,     // [16,32]
                                  const float* __restrict__ att_s,    // [3,32]
                                  const float* __restrict__ att_r,    // [1,32]
                                  const float* __restrict__ XL,       // [N,128]
                                  const float* __restrict__ XR,       // [N,128]
                                  const float* __restrict__ ef_loop,  // [128]
                                  float* __restrict__ score,          // [E+N,4]
                                  float* __restrict__ smax,           // [N,4]
                                  long long E, int N) {
    __shared__ float s_wes[EE_OUT * 96];
    __shared__ float s_wer[EE_OUT * 32];
    __shared__ float s_att[128];   // 96 std + 32 rep
    __shared__ float s_efl[128];
    for (int i = threadIdx.x; i < EE_OUT * 96; i += blockDim.x) s_wes[i] = we_s[i];
    for (int i = threadIdx.x; i < EE_OUT * 32; i += blockDim.x) s_wer[i] = we_r[i];
    for (int i = threadIdx.x; i < 96;  i += blockDim.x) s_att[i]      = att_s[i];
    for (int i = threadIdx.x; i < 32;  i += blockDim.x) s_att[96 + i] = att_r[i];
    for (int i = threadIdx.x; i < 128; i += blockDim.x) s_efl[i]      = ef_loop[i];
    __syncthreads();

    const long long total  = (E + (long long)N) * H_ALL;
    const long long stride = (long long)gridDim.x * blockDim.x;
    for (long long idx = (long long)blockIdx.x * blockDim.x + threadIdx.x;
         idx < total; idx += stride) {
        const long long e = idx >> 2;
        const int h = (int)(idx & 3);
        const int coff = (h < 3) ? h * C_DIM : 96;

        int s, d;
        bool live = true;
        if (e < E) {
            s = src[e]; d = dst[e];
            if (h == 3 && mask[e] == 0) live = false;      // repulsion subset
        } else {
            s = d = (int)(e - E);                           // self loop
        }

        float sc;
        if (!live) {
            sc = -INFINITY;
        } else {
            float eav[EE_OUT];
            if (e < E) {
                const float4* ea4 = (const float4*)(ea + e * EE_OUT);
                #pragma unroll
                for (int q = 0; q < EE_OUT / 4; ++q) {
                    const float4 v = ea4[q];
                    eav[4 * q] = v.x; eav[4 * q + 1] = v.y;
                    eav[4 * q + 2] = v.z; eav[4 * q + 3] = v.w;
                }
            }
            const float4* xl4 = (const float4*)(XL + (size_t)s * (H_ALL * C_DIM) + coff);
            const float4* xr4 = (const float4*)(XR + (size_t)d * (H_ALL * C_DIM) + coff);
            const float* att = s_att + coff;
            sc = 0.0f;
            #pragma unroll
            for (int c4 = 0; c4 < C_DIM / 4; ++c4) {
                const float4 va = xl4[c4];
                const float4 vb = xr4[c4];
                const float comp[4] = {va.x + vb.x, va.y + vb.y, va.z + vb.z, va.w + vb.w};
                #pragma unroll
                for (int u = 0; u < 4; ++u) {
                    const int c = 4 * c4 + u;
                    float ef;
                    if (e < E) {
                        ef = 0.0f;
                        if (h < 3) {
                            #pragma unroll
                            for (int j = 0; j < EE_OUT; ++j)
                                ef += eav[j] * s_wes[j * 96 + coff + c];
                        } else {
                            #pragma unroll
                            for (int j = 0; j < EE_OUT; ++j)
                                ef += eav[j] * s_wer[j * 32 + c];
                        }
                    } else {
                        ef = s_efl[coff + c];
                    }
                    float m = comp[u] + ef;
                    m = (m > 0.0f) ? m : NEG_SLOPE * m;     // leaky relu
                    sc += m * att[c];
                }
            }
            atomicMaxFloat(&smax[(size_t)d * H_ALL + h], sc);
        }
        score[idx] = sc;
    }
}

// ---------------------------------------------------------------------------
// K5: a = exp(score - smax[dst]); atomic segment sum -> denom
// ---------------------------------------------------------------------------
__global__ void softmax_num_kernel(const int* __restrict__ dst,
                                   const float* __restrict__ smax,
                                   float* __restrict__ score,   // in: score, out: a
                                   float* __restrict__ denom,
                                   long long E, int N) {
    const long long total  = (E + (long long)N) * H_ALL;
    const long long stride = (long long)gridDim.x * blockDim.x;
    for (long long idx = (long long)blockIdx.x * blockDim.x + threadIdx.x;
         idx < total; idx += stride) {
        const long long e = idx >> 2;
        const int h = (int)(idx & 3);
        const int d = (e < E) ? dst[e] : (int)(e - E);
        const float sc = score[idx];
        float a = 0.0f;
        if (sc != -INFINITY) a = __expf(sc - smax[(size_t)d * H_ALL + h]);
        score[idx] = a;
        if (a != 0.0f) atomicAdd(&denom[(size_t)d * H_ALL + h], a);
    }
}

// ---------------------------------------------------------------------------
// K6: out[dst, head] += (a / denom[dst]) * xl[src, head]
// ---------------------------------------------------------------------------
__global__ void aggregate_kernel(const int* __restrict__ src,
                                 const int* __restrict__ dst,
                                 const float* __restrict__ score,   // a
                                 const float* __restrict__ denom,
                                 const float* __restrict__ XL,
                                 float* __restrict__ out,
                                 long long E, int N) {
    const long long total  = (E + (long long)N) * H_ALL;
    const long long stride = (long long)gridDim.x * blockDim.x;
    for (long long idx = (long long)blockIdx.x * blockDim.x + threadIdx.x;
         idx < total; idx += stride) {
        const float a = score[idx];
        if (a == 0.0f) continue;
        const long long e = idx >> 2;
        const int h = (int)(idx & 3);
        const int coff = (h < 3) ? h * C_DIM : 96;
        int s, d;
        if (e < E) { s = src[e]; d = dst[e]; } else { s = d = (int)(e - E); }
        const float w = a / denom[(size_t)d * H_ALL + h];
        const float4* xl4 = (const float4*)(XL + (size_t)s * (H_ALL * C_DIM) + coff);
        __builtin_prefetch(xl4, 0, 1);                      // global_prefetch_b8
        float* orow = out + (size_t)d * (H_ALL * C_DIM) + coff;
        #pragma unroll
        for (int c4 = 0; c4 < C_DIM / 4; ++c4) {
            const float4 v = xl4[c4];
            atomicAdd(&orow[4 * c4],     w * v.x);
            atomicAdd(&orow[4 * c4 + 1], w * v.y);
            atomicAdd(&orow[4 * c4 + 2], w * v.z);
            atomicAdd(&orow[4 * c4 + 3], w * v.w);
        }
    }
}

// ---------------------------------------------------------------------------
extern "C" void kernel_launch(void* const* d_in, const int* in_sizes, int n_in,
                              void* d_out, int out_size, void* d_ws, size_t ws_size,
                              hipStream_t stream) {
    // Inputs in setup_inputs() dict order
    const float*         x      = (const float*)d_in[0];
    const float*         raw    = (const float*)d_in[1];
    const int*           eidx   = (const int*)d_in[2];     // [2, E]
    const unsigned char* stmask = (const unsigned char*)d_in[3];
    const float* ee_w1 = (const float*)d_in[4];
    const float* ee_b1 = (const float*)d_in[5];
    const float* ee_w2 = (const float*)d_in[6];
    const float* ee_b2 = (const float*)d_in[7];
    const float* wl_s  = (const float*)d_in[8];
    const float* bl_s  = (const float*)d_in[9];
    const float* wr_s  = (const float*)d_in[10];
    const float* br_s  = (const float*)d_in[11];
    const float* we_s  = (const float*)d_in[12];
    const float* att_s = (const float*)d_in[13];
    const float* bias_s= (const float*)d_in[14];
    const float* wl_r  = (const float*)d_in[15];
    const float* bl_r  = (const float*)d_in[16];
    const float* wr_r  = (const float*)d_in[17];
    const float* br_r  = (const float*)d_in[18];
    const float* we_r  = (const float*)d_in[19];
    const float* att_r = (const float*)d_in[20];
    const float* bias_r= (const float*)d_in[21];

    const int       N = in_sizes[0] / IN_DIM;
    const long long E = (long long)in_sizes[3];
    const int*      src = eidx;
    const int*      dst = eidx + E;
    float*          out = (float*)d_out;

    // --- workspace carve ---
    float* ws = (float*)d_ws;
    size_t o = 0;
    float* XL    = ws + o; o += (size_t)N * (H_ALL * C_DIM);
    float* XR    = ws + o; o += (size_t)N * (H_ALL * C_DIM);
    float* ea    = ws + o; o += (size_t)E * EE_OUT;
    float* score = ws + o; o += ((size_t)E + N) * H_ALL;
    float* smax  = ws + o; o += (size_t)N * H_ALL;
    float* denom = ws + o; o += (size_t)N * H_ALL;
    float* accum = ws + o; o += 64;       // 16 sum + 16 sum_rep + 1 cnt
    float* efl   = ws + o; o += 128;      // self-loop ef (96 std + 32 rep)
    // bf16 region (2-byte elems carved from the same pool, 4B-aligned)
    __bf16* bfbase = (__bf16*)(ws + o);
    size_t ob = 0;
    __bf16* Xb    = bfbase + ob; ob += (size_t)N * IN_DIM;
    __bf16* Wt_ls = bfbase + ob; ob += (size_t)96 * IN_DIM;
    __bf16* Wt_lr = bfbase + ob; ob += (size_t)32 * IN_DIM;
    __bf16* Wt_rs = bfbase + ob; ob += (size_t)96 * IN_DIM;
    __bf16* Wt_rr = bfbase + ob; ob += (size_t)32 * IN_DIM;

    const long long totalEH = (E + (long long)N) * H_ALL;
    const int  tpb   = 256;
    const int  ginit = (int)min((long long)2048, ((long long)N + tpb - 1) / tpb);
    const int  gcvt  = (int)min((long long)2048, ((long long)N * IN_DIM + tpb - 1) / tpb);
    const int  gedge = (int)min((long long)8192, (E + tpb - 1) / tpb);
    const int  geh   = (int)min((long long)16384, (totalEH + tpb - 1) / tpb);
    const int  mtile = (N + 15) / 16;

    // K0: init (must precede encoder/score passes)
    init_kernel<<<ginit, tpb, 0, stream>>>(out, bias_s, bias_r, smax, denom, accum, N);

    // Pre-convert operands to bf16 (W transposed for contiguous B fragments)
    cvt_x_bf16<<<gcvt, tpb, 0, stream>>>(x, Xb, (long long)N * IN_DIM);
    cvt_w_bf16_t<<<48, tpb, 0, stream>>>(wl_s, Wt_ls, IN_DIM, 96);
    cvt_w_bf16_t<<<16, tpb, 0, stream>>>(wl_r, Wt_lr, IN_DIM, 32);
    cvt_w_bf16_t<<<48, tpb, 0, stream>>>(wr_s, Wt_rs, IN_DIM, 96);
    cvt_w_bf16_t<<<16, tpb, 0, stream>>>(wr_r, Wt_rr, IN_DIM, 32);

    // K1: node transforms (WMMA bf16, f32 accumulate)
    gemm_wmma_bf16<<<dim3(mtile, 6), 32, 0, stream>>>(Xb, Wt_ls, bl_s, XL, N, 96, 128, 0);
    gemm_wmma_bf16<<<dim3(mtile, 2), 32, 0, stream>>>(Xb, Wt_lr, bl_r, XL, N, 32, 128, 96);
    gemm_wmma_bf16<<<dim3(mtile, 6), 32, 0, stream>>>(Xb, Wt_rs, br_s, XR, N, 96, 128, 0);
    gemm_wmma_bf16<<<dim3(mtile, 2), 32, 0, stream>>>(Xb, Wt_rr, br_r, XR, N, 32, 128, 96);

    // K2: edge encoder + mean accumulation
    edge_encoder_kernel<<<gedge, tpb, 0, stream>>>(raw, ee_w1, ee_b1, ee_w2, ee_b2,
                                                   stmask, ea, accum, E);
    // K3: means + self-loop ef
    finalize_means_kernel<<<1, 128, 0, stream>>>(accum, we_s, we_r, efl, E);

    // K4: scores + segment max
    edge_score_kernel<<<geh, tpb, 0, stream>>>(src, dst, stmask, ea, we_s, we_r,
                                               att_s, att_r, XL, XR, efl,
                                               score, smax, E, N);
    // K5: exp + segment sum
    softmax_num_kernel<<<geh, tpb, 0, stream>>>(dst, smax, score, denom, E, N);
    // K6: normalized scatter-add
    aggregate_kernel<<<geh, tpb, 0, stream>>>(src, dst, score, denom, XL, out, E, N);
}
